// model_mamba_36301063586031
// MI455X (gfx1250) — matrix operand order repaired and checked by reading, not compiled, not run on hardware
//
#include <hip/hip_runtime.h>
#include <hip/hip_bf16.h>

// ---------------------------------------------------------------------------
// Model dims (compile-time constants from the reference)
// ---------------------------------------------------------------------------
#define BB       4
#define LL       2048
#define D_MODEL  256
#define D_INNER  512
#define D_STATE  16
#define D_CONV   4
#define DT_RANK  16
#define OUT_DIM  128
#define M_TOK    (BB * LL)          // 8192 rows everywhere

typedef __attribute__((ext_vector_type(16))) _Float16 v16h;
typedef __attribute__((ext_vector_type(8)))  _Float16 v8h;
typedef __attribute__((ext_vector_type(8)))  float    v8f;

enum { EPI_NONE = 0, EPI_BIAS = 1, EPI_BIAS_RELU = 2, EPI_BIAS_SOFTPLUS = 3 };

// ---------------------------------------------------------------------------
// Operand packing: f32 -> f16 with zero padding to Kp (mult of 32) columns
// and (for weights) Np (mult of 64) rows. This removes every bounds check
// from the WMMA GEMM hot loop and makes all fragment loads contiguous b128.
// ---------------------------------------------------------------------------
__global__ __launch_bounds__(256)
void pack_a_f16(const float* __restrict__ A, int lda, int K,
                _Float16* __restrict__ Ah, int Kp, int total)
{
    int i = blockIdx.x * blockDim.x + threadIdx.x;
    if (i >= total) return;
    const int k = i % Kp;
    const int m = i / Kp;
    Ah[i] = (k < K) ? (_Float16)A[(size_t)m * lda + k] : (_Float16)0.0f;
}

__global__ __launch_bounds__(256)
void pack_w_f16(const float* __restrict__ W, int ldw, int N, int K,
                _Float16* __restrict__ Wh, int Kp, int total)
{
    int i = blockIdx.x * blockDim.x + threadIdx.x;
    if (i >= total) return;
    const int k = i % Kp;
    const int n = i / Kp;
    Wh[i] = (n < N && k < K) ? (_Float16)W[(size_t)n * ldw + k] : (_Float16)0.0f;
}

// ---------------------------------------------------------------------------
// WMMA GEMM on packed f16 operands:
//   C[m,n] = epi( sum_k Ah[m,k] * Wh[n,k] + bias[n] ),  m in [0,M_TOK)
// Ah: M_TOK x KP,  Wh: Np x KP (Np mult of 64, zero padded),  C: f32, stride ldc.
// 128 threads = 4 waves per block; each wave computes a 16x64 C strip
// (4 accumulators) so one A fragment feeds 4 v_wmma per k-step:
// 2.5 b128 loads per wmma instead of 4, and 4x less A traffic.
// KP is a template parameter -> fully unrolled wmma chain.
// Fragments use contiguous k order (K = k0 + half_sel*16 + s) in both A and B.
// ---------------------------------------------------------------------------
template<int KP, int EPI>
__global__ __launch_bounds__(128)
void gemm_wmma(const _Float16* __restrict__ Ah,
               const _Float16* __restrict__ Wh,
               const float* __restrict__ bias,
               float* __restrict__ C, int ldc, int N, int strips_n)
{
    const int wave  = threadIdx.x >> 5;
    const int strip = blockIdx.x * 4 + wave;
    const int tm    = strip / strips_n;
    const int tn    = strip - tm * strips_n;
    const int m0    = tm << 4;
    const int n0    = tn << 6;            // 64-wide strip

    const int lane     = threadIdx.x & 31;
    const int half_sel = lane >> 4;       // 0: lanes 0-15, 1: lanes 16-31
    const int l15      = lane & 15;

    const _Float16* __restrict__ pa = Ah + (size_t)(m0 + l15) * KP;   // A row
    const _Float16* __restrict__ pw0 = Wh + (size_t)(n0 +  0 + l15) * KP;
    const _Float16* __restrict__ pw1 = Wh + (size_t)(n0 + 16 + l15) * KP;
    const _Float16* __restrict__ pw2 = Wh + (size_t)(n0 + 32 + l15) * KP;
    const _Float16* __restrict__ pw3 = Wh + (size_t)(n0 + 48 + l15) * KP;

    v8f acc0 = {}, acc1 = {}, acc2 = {}, acc3 = {};

#pragma unroll
    for (int k0 = 0; k0 < KP; k0 += 32) {
        // A 16x32 f16 fragment: lane holds K = k0 + half_sel*16 + {0..15}
        union { v16h v; v8h h[2]; } a;
        a.h[0] = *(const v8h*)(pa + k0 + (half_sel << 4));
        a.h[1] = *(const v8h*)(pa + k0 + (half_sel << 4) + 8);
        // B 32x16 f16 fragments: lane holds K = k0 + half_sel*16 + {0..15}
        const int kb = k0 + (half_sel << 4);
        const v16h b0 = *(const v16h*)(pw0 + kb);
        const v16h b1 = *(const v16h*)(pw1 + kb);
        const v16h b2 = *(const v16h*)(pw2 + kb);
        const v16h b3 = *(const v16h*)(pw3 + kb);
        acc0 = __builtin_amdgcn_wmma_f32_16x16x32_f16(false, a.v, false, b0,
                                                      (short)0, acc0, false, false);
        acc1 = __builtin_amdgcn_wmma_f32_16x16x32_f16(false, a.v, false, b1,
                                                      (short)0, acc1, false, false);
        acc2 = __builtin_amdgcn_wmma_f32_16x16x32_f16(false, a.v, false, b2,
                                                      (short)0, acc2, false, false);
        acc3 = __builtin_amdgcn_wmma_f32_16x16x32_f16(false, a.v, false, b3,
                                                      (short)0, acc3, false, false);
    }

    // Epilogue: C 16x16 f32 layout per tile: M = m0 + r + half_sel*8, N = col
    v8f accs[4] = { acc0, acc1, acc2, acc3 };
#pragma unroll
    for (int j = 0; j < 4; ++j) {
        const int bcol = n0 + (j << 4) + l15;
        if (bcol < N) {
            const float bi = (EPI != EPI_NONE) ? bias[bcol] : 0.0f;
#pragma unroll
            for (int r = 0; r < 8; ++r) {
                const int m = m0 + r + (half_sel << 3);
                float v = accs[j][r];
                if (EPI != EPI_NONE)          v += bi;
                if (EPI == EPI_BIAS_RELU)     v = v > 0.0f ? v : 0.0f;
                if (EPI == EPI_BIAS_SOFTPLUS) v = (v > 20.0f) ? v : logf(1.0f + __expf(v));
                C[(size_t)m * ldc + bcol] = v;
            }
        }
    }
}

// ---------------------------------------------------------------------------
// Causal depthwise conv1d (pad left D_CONV-1) + bias + SiLU.
// Reads the xs half (cols [0,512)) of the xz buffer (row stride 2*D_INNER).
// ---------------------------------------------------------------------------
__global__ __launch_bounds__(256)
void conv_silu_kernel(const float* __restrict__ xz,
                      const float* __restrict__ conv_w,
                      const float* __restrict__ conv_b,
                      float* __restrict__ xc, int total)
{
    int i = blockIdx.x * blockDim.x + threadIdx.x;
    if (i >= total) return;
    const int d  = i % D_INNER;
    const int bl = i / D_INNER;
    const int l  = bl % LL;
    const int b  = bl / LL;

    float s = conv_b[d];
#pragma unroll
    for (int j = 0; j < D_CONV; ++j) {
        const int ls = l - (D_CONV - 1) + j;
        if (ls >= 0)
            s += conv_w[d * D_CONV + j] *
                 xz[((size_t)b * LL + ls) * (2 * D_INNER) + d];
    }
    xc[i] = s / (1.0f + __expf(-s));   // SiLU
}

// ---------------------------------------------------------------------------
// Selective scan. One workgroup per batch element, one thread per d_inner
// channel; 16 SSM states per thread live in registers. Bm/Cm for the current
// timestep are broadcast through LDS. Output fused with y * silu(z).
// ---------------------------------------------------------------------------
__global__ __launch_bounds__(D_INNER)
void scan_kernel(const float* __restrict__ dt,
                 const float* __restrict__ xdbl,
                 const float* __restrict__ xc,
                 const float* __restrict__ xz,
                 const float* __restrict__ A_log,
                 const float* __restrict__ Dvec,
                 float* __restrict__ y)
{
    __shared__ float sB[D_STATE];
    __shared__ float sC[D_STATE];

    const int b = blockIdx.x;
    const int d = threadIdx.x;

    float Arow[D_STATE];
#pragma unroll
    for (int n = 0; n < D_STATE; ++n)
        Arow[n] = -__expf(A_log[d * D_STATE + n]);

    float h[D_STATE];
#pragma unroll
    for (int n = 0; n < D_STATE; ++n) h[n] = 0.0f;

    const float Dd = Dvec[d];

    for (int l = 0; l < LL; ++l) {
        const size_t row = (size_t)b * LL + l;
        if (threadIdx.x < 2 * D_STATE) {
            const int   t = threadIdx.x;
            const float v = xdbl[row * (DT_RANK + 2 * D_STATE) + DT_RANK + t];
            if (t < D_STATE) sB[t] = v; else sC[t - D_STATE] = v;
        }
        __syncthreads();

        const float dtv = dt[row * D_INNER + d];
        const float u   = xc[row * D_INNER + d];
        float yv = 0.0f;
#pragma unroll
        for (int n = 0; n < D_STATE; ++n) {
            const float dA = __expf(dtv * Arow[n]);
            h[n] = dA * h[n] + dtv * sB[n] * u;
            yv  += h[n] * sC[n];
        }
        yv += Dd * u;

        const float z  = xz[row * (2 * D_INNER) + D_INNER + d];
        const float sz = z / (1.0f + __expf(-z));
        y[row * D_INNER + d] = yv * sz;

        __syncthreads();
    }
}

// ---------------------------------------------------------------------------
// Host-side helpers
// ---------------------------------------------------------------------------
static inline void launch_pack_a(const float* A, int lda, int K,
                                 _Float16* Ah, int Kp, hipStream_t s)
{
    const int total = M_TOK * Kp;
    hipLaunchKernelGGL(pack_a_f16, dim3((total + 255) / 256), dim3(256), 0, s,
                       A, lda, K, Ah, Kp, total);
}

static inline void launch_pack_w(const float* W, int ldw, int N, int K,
                                 _Float16* Wh, int Kp, hipStream_t s)
{
    const int Np    = (N + 63) & ~63;     // pad rows to full 64-wide strips
    const int total = Np * Kp;
    hipLaunchKernelGGL(pack_w_f16, dim3((total + 255) / 256), dim3(256), 0, s,
                       W, ldw, N, K, Wh, Kp, total);
}

template<int KP, int EPI>
static inline void launch_gemm(const _Float16* Ah, const _Float16* Wh,
                               const float* bias, float* C, int ldc, int N,
                               hipStream_t s)
{
    const int strips_n = (N + 63) >> 6;
    const int strips   = (M_TOK >> 4) * strips_n;   // tiles_m = 512 -> mult of 4
    hipLaunchKernelGGL((gemm_wmma<KP, EPI>), dim3(strips / 4), dim3(128), 0, s,
                       Ah, Wh, bias, C, ldc, N, strips_n);
}

extern "C" void kernel_launch(void* const* d_in, const int* in_sizes, int n_in,
                              void* d_out, int out_size, void* d_ws, size_t ws_size,
                              hipStream_t stream)
{
    const float* x       = (const float*)d_in[0];
    const float* W_in    = (const float*)d_in[1];
    const float* conv_w  = (const float*)d_in[2];
    const float* conv_b  = (const float*)d_in[3];
    const float* W_xproj = (const float*)d_in[4];
    const float* W_dt    = (const float*)d_in[5];
    const float* b_dt    = (const float*)d_in[6];
    const float* A_log   = (const float*)d_in[7];
    const float* Dvec    = (const float*)d_in[8];
    const float* W_out   = (const float*)d_in[9];
    const float* fc1_w0  = (const float*)d_in[10];
    const float* fc1_b0  = (const float*)d_in[11];
    const float* fc1_w1  = (const float*)d_in[12];
    const float* fc1_b1  = (const float*)d_in[13];
    const float* fc1_w2  = (const float*)d_in[14];
    const float* fc1_b2  = (const float*)d_in[15];
    const float* fc2_w   = (const float*)d_in[16];
    const float* fc2_b   = (const float*)d_in[17];

    // ---- fp32 intermediates -------------------------------------------------
    float* ws   = (float*)d_ws;
    float* xz   = ws;  ws += (size_t)M_TOK * 2 * D_INNER;            // 8192x1024
    float* xc   = ws;  ws += (size_t)M_TOK * D_INNER;                // 8192x512
    float* xdbl = ws;  ws += (size_t)M_TOK * (DT_RANK + 2*D_STATE);  // 8192x48
    float* dt   = ws;  ws += (size_t)M_TOK * D_INNER;                // 8192x512
    float* y    = ws;  ws += (size_t)M_TOK * D_INNER;                // 8192x512
    float* mout = ws;  ws += (size_t)M_TOK * D_MODEL;                // 8192x256
    float* h1   = ws;  ws += (size_t)M_TOK * 100;                    // 8192x100
    float* h2   = ws;  ws += (size_t)M_TOK * 100;                    // 8192x100
    float* h3   = ws;  ws += (size_t)M_TOK * OUT_DIM;                // 8192x128

    // ---- packed f16 operand buffers (weights padded to Np mult of 64) ------
    _Float16* hp = (_Float16*)ws;
    _Float16* Ah      = hp;  hp += (size_t)M_TOK * 512;   // activation pack (max Kp=512)
    _Float16* Wh_in   = hp;  hp += (size_t)1024 * 256;
    _Float16* Wh_xprj = hp;  hp += (size_t)64   * 512;
    _Float16* Wh_dt   = hp;  hp += (size_t)512  * 32;
    _Float16* Wh_out  = hp;  hp += (size_t)256  * 512;
    _Float16* Wh_f10  = hp;  hp += (size_t)128  * 256;
    _Float16* Wh_f11  = hp;  hp += (size_t)128  * 128;
    _Float16* Wh_f12  = hp;  hp += (size_t)128  * 128;
    _Float16* Wh_f2   = hp;  hp += (size_t)128  * 128;

    float* out = (float*)d_out;                                      // 8192x128

    // ---- pack all weights once (independent, small) ------------------------
    launch_pack_w(W_in,    D_MODEL, 2*D_INNER,          D_MODEL, Wh_in,   256, stream);
    launch_pack_w(W_xproj, D_INNER, DT_RANK+2*D_STATE,  D_INNER, Wh_xprj, 512, stream);
    launch_pack_w(W_dt,    DT_RANK, D_INNER,            DT_RANK, Wh_dt,   32,  stream);
    launch_pack_w(W_out,   D_INNER, D_MODEL,            D_INNER, Wh_out,  512, stream);
    launch_pack_w(fc1_w0,  D_MODEL, 100,                D_MODEL, Wh_f10,  256, stream);
    launch_pack_w(fc1_w1,  100,     100,                100,     Wh_f11,  128, stream);
    launch_pack_w(fc1_w2,  100,     OUT_DIM,            100,     Wh_f12,  128, stream);
    launch_pack_w(fc2_w,   OUT_DIM, OUT_DIM,            OUT_DIM, Wh_f2,   128, stream);

    // 1) xz = x @ W_in^T                                  (8192 x 1024 x 256)
    launch_pack_a(x, D_MODEL, D_MODEL, Ah, 256, stream);
    launch_gemm<256, EPI_NONE>(Ah, Wh_in, nullptr, xz, 2*D_INNER, 2*D_INNER, stream);

    // 2) causal depthwise conv + bias + SiLU -> xc
    {
        const int total = M_TOK * D_INNER;
        hipLaunchKernelGGL(conv_silu_kernel, dim3((total + 255) / 256), dim3(256),
                           0, stream, xz, conv_w, conv_b, xc, total);
    }

    // 3) x_dbl = xc @ W_xproj^T                           (8192 x 48 x 512)
    launch_pack_a(xc, D_INNER, D_INNER, Ah, 512, stream);
    launch_gemm<512, EPI_NONE>(Ah, Wh_xprj, nullptr, xdbl,
                               DT_RANK + 2*D_STATE, DT_RANK + 2*D_STATE, stream);

    // 4) dt = softplus(x_dbl[:, :16] @ W_dt^T + b_dt)     (8192 x 512 x 16)
    launch_pack_a(xdbl, DT_RANK + 2*D_STATE, DT_RANK, Ah, 32, stream);
    launch_gemm<32, EPI_BIAS_SOFTPLUS>(Ah, Wh_dt, b_dt, dt, D_INNER, D_INNER, stream);

    // 5) selective scan (+ D*xc, * silu(z)) -> y
    hipLaunchKernelGGL(scan_kernel, dim3(BB), dim3(D_INNER), 0, stream,
                       dt, xdbl, xc, xz, A_log, Dvec, y);

    // 6) mout = y @ W_out^T                               (8192 x 256 x 512)
    launch_pack_a(y, D_INNER, D_INNER, Ah, 512, stream);
    launch_gemm<512, EPI_NONE>(Ah, Wh_out, nullptr, mout, D_MODEL, D_MODEL, stream);

    // 7) MLP
    launch_pack_a(mout, D_MODEL, D_MODEL, Ah, 256, stream);
    launch_gemm<256, EPI_BIAS_RELU>(Ah, Wh_f10, fc1_b0, h1, 100, 100, stream);

    launch_pack_a(h1, 100, 100, Ah, 128, stream);
    launch_gemm<128, EPI_BIAS_RELU>(Ah, Wh_f11, fc1_b1, h2, 100, 100, stream);

    launch_pack_a(h2, 100, 100, Ah, 128, stream);
    launch_gemm<128, EPI_BIAS>(Ah, Wh_f12, fc1_b2, h3, OUT_DIM, OUT_DIM, stream);

    // 8) out = h3 @ fc2_w^T + fc2_b                       (8192 x 128 x 128)
    launch_pack_a(h3, OUT_DIM, OUT_DIM, Ah, 128, stream);
    launch_gemm<128, EPI_BIAS>(Ah, Wh_f2, fc2_b, out, OUT_DIM, OUT_DIM, stream);
}
